// Pointnet2_msg_8323646620001
// MI455X (gfx1250) — compile-verified, matrix operand order checked by reading, not run on hardware
//
#include <hip/hip_runtime.h>

typedef __attribute__((ext_vector_type(16))) _Float16 v16h;
typedef __attribute__((ext_vector_type(8)))  _Float16 v8h;
typedef __attribute__((ext_vector_type(8)))  float    v8f;

#define THREADS 256

// ---------------- transpose (B,3,N) -> (B,N,3) ----------------
__global__ void transpose_xyz_k(const float* __restrict__ in, float* __restrict__ out,
                                int B, int N) {
  int i = blockIdx.x * blockDim.x + threadIdx.x;
  int total = B * N * 3;
  if (i >= total) return;
  int b = i / (N * 3), rem = i % (N * 3), n = rem / 3, c = rem % 3;
  out[i] = in[((size_t)b * 3 + c) * N + n];
}

// ---------------- farthest point sampling: one block per batch ----------------
__global__ void fps_k(const float* __restrict__ xyz, int N, int npoint,
                      int* __restrict__ fidx, float* __restrict__ nxyz,
                      float* __restrict__ dist) {
  int b = blockIdx.x, t = threadIdx.x;
  const float* pts = xyz + (size_t)b * N * 3;
  float* d = dist + (size_t)b * N;
  __shared__ float sv[THREADS];
  __shared__ int   si[THREADS];
  for (int i = t; i < N; i += THREADS) d[i] = 1e10f;
  __syncthreads();
  int far = 0;
  for (int it = 0; it < npoint; ++it) {
    float cx = pts[far * 3 + 0], cy = pts[far * 3 + 1], cz = pts[far * 3 + 2];
    if (t == 0) {
      fidx[b * npoint + it] = far;
      nxyz[((size_t)b * npoint + it) * 3 + 0] = cx;
      nxyz[((size_t)b * npoint + it) * 3 + 1] = cy;
      nxyz[((size_t)b * npoint + it) * 3 + 2] = cz;
    }
    float best = -1.f; int bi = 0;
    for (int i = t; i < N; i += THREADS) {
      float dx = pts[i * 3 + 0] - cx, dy = pts[i * 3 + 1] - cy, dz = pts[i * 3 + 2] - cz;
      float dd = dx * dx + dy * dy + dz * dz;
      float dm = fminf(d[i], dd);
      d[i] = dm;
      if (dm > best) { best = dm; bi = i; }
    }
    sv[t] = best; si[t] = bi;
    __syncthreads();
    for (int s = THREADS / 2; s > 0; s >>= 1) {
      if (t < s) {
        if (sv[t + s] > sv[t] || (sv[t + s] == sv[t] && si[t + s] < si[t])) {
          sv[t] = sv[t + s]; si[t] = si[t + s];
        }
      }
      __syncthreads();
    }
    far = si[0];
    __syncthreads();
  }
}

// ---------------- ball query: one wave32 per (batch,centroid), ordered first-k ----------------
__global__ void query_ball_k(const float* __restrict__ xyz, const float* __restrict__ nxyz,
                             int N, int np, float r2, int k, int* __restrict__ gidx) {
  int g = blockIdx.x;
  int lane = threadIdx.x;
  int b = g / np;
  const float* pts = xyz + (size_t)b * N * 3;
  float cx = nxyz[(size_t)g * 3 + 0], cy = nxyz[(size_t)g * 3 + 1], cz = nxyz[(size_t)g * 3 + 2];
  int* out = gidx + (size_t)g * k;
  int cnt = 0, first = 0;
  for (int base = 0; base < N && cnt < k; base += 32) {
    int i = base + lane;
    bool hit = false;
    if (i < N) {
      float dx = pts[i * 3 + 0] - cx, dy = pts[i * 3 + 1] - cy, dz = pts[i * 3 + 2] - cz;
      hit = (dx * dx + dy * dy + dz * dz) <= r2;
    }
    unsigned mask = (unsigned)__ballot(hit);   // wave32: low 32 bits valid
    if (cnt == 0 && mask) first = base + __ffs((int)mask) - 1;
    int pre = __popc(mask & ((1u << lane) - 1u));
    if (hit && (cnt + pre) < k) out[cnt + pre] = i;
    cnt += __popc(mask);
  }
  if (cnt > k) cnt = k;
  for (int j = cnt + lane; j < k; j += 32) out[j] = first;
}

// ---------------- gather grouped features into f16 X[rows, Kpad] ----------------
__global__ void gather_k(const float* __restrict__ xyz, const float* __restrict__ nxyz,
                         const int* __restrict__ gidx, const _Float16* __restrict__ feat,
                         int Cf, int N, int np, int k, int Kpad, int rows,
                         _Float16* __restrict__ X) {
  int r = blockIdx.x * blockDim.x + threadIdx.x;
  if (r >= rows) return;
  int b = r / (np * k);
  int pg = (r / k) % np;
  int idx = gidx[r];
  const float* P  = xyz  + ((size_t)b * N  + idx) * 3;
  const float* Cc = nxyz + ((size_t)b * np + pg) * 3;
  _Float16* x = X + (size_t)r * Kpad;
  int o = 0;
  if (feat) {
    const _Float16* f = feat + ((size_t)b * N + idx) * Cf;
    for (int c = 0; c < Cf; c++) x[o++] = f[c];
  }
  x[o++] = (_Float16)(P[0] - Cc[0]);
  x[o++] = (_Float16)(P[1] - Cc[1]);
  x[o++] = (_Float16)(P[2] - Cc[2]);
  for (; o < Kpad; o++) x[o] = (_Float16)0.f;
}

// ---------------- build SA3 input: concat(xyz, pts) -> X[rows, Kpad] ----------------
__global__ void build_sa3_k(const float* __restrict__ xyz, const _Float16* __restrict__ pts,
                            int C, int Kpad, int rows, _Float16* __restrict__ X) {
  int r = blockIdx.x * blockDim.x + threadIdx.x;
  if (r >= rows) return;
  _Float16* x = X + (size_t)r * Kpad;
  x[0] = (_Float16)xyz[(size_t)r * 3 + 0];
  x[1] = (_Float16)xyz[(size_t)r * 3 + 1];
  x[2] = (_Float16)xyz[(size_t)r * 3 + 2];
  const _Float16* f = pts + (size_t)r * C;
  for (int c = 0; c < C; c++) x[3 + c] = f[c];
  for (int o = 3 + C; o < Kpad; o++) x[o] = (_Float16)0.f;
}

// ---------------- fold BN into f16 weights/bias: w16[Npad,Kpad], b16[Npad] ----------------
__global__ void fold_bn_k(const float* __restrict__ w, const float* __restrict__ b,
                          const float* __restrict__ g, const float* __restrict__ be,
                          int Co, int Ci, int Kpad, int Npad,
                          _Float16* __restrict__ w16, _Float16* __restrict__ b16) {
  int i = blockIdx.x * blockDim.x + threadIdx.x;
  int total = Npad * Kpad;
  if (i >= total) return;
  int o = i / Kpad, c = i % Kpad;
  float v = 0.f;
  if (o < Co && c < Ci) {
    float s = g ? g[o] * rsqrtf(1.f + 1e-5f) : 1.f;
    v = w[(size_t)o * Ci + c] * s;
  }
  w16[i] = (_Float16)v;
  if (c == 0) {
    float bv = 0.f;
    if (o < Co) {
      float s = g ? g[o] * rsqrtf(1.f + 1e-5f) : 1.f;
      bv = b[o] * s + (be ? be[o] : 0.f);
    }
    b16[o] = (_Float16)bv;
  }
}

// ---------------- fragment load: CDNA5 16-bit A/B layout ----------------
// lane l -> matrix row (l&15); halves 0..7 at K = kb + 8*(l>>4),
// halves 8..15 at K = kb + 16 + 8*(l>>4). Two aligned 16B loads.
__device__ __forceinline__ v16h ldfrag(const _Float16* p) {
  v8h lo = *(const v8h*)p;
  v8h hi = *(const v8h*)(p + 16);
  v16h f;
#pragma unroll
  for (int i = 0; i < 8; i++) { f[i] = lo[i]; f[8 + i] = hi[i]; }
  return f;
}

// ---------------- WMMA GEMM: Y = act(X[M,K] * W[Npad,K]^T + bias) ----------------
// One wave32 computes a 32x32 output tile (2x2 of 16x16 WMMA tiles):
// 8 b128 loads feed 4 v_wmma per k-step. Out-of-range rows/cols are CLAMPED to
// the last valid row (duplicate compute, never stored) -> no EXEC divergence.
__global__ void gemm_bias_act_k(const _Float16* __restrict__ X,
                                const _Float16* __restrict__ W,
                                const _Float16* __restrict__ Bv,
                                _Float16* __restrict__ Yh, float* __restrict__ Yf,
                                int M, int K, int N, int Npad, int ldy, int relu) {
  int lane = threadIdx.x;
  int half = lane >> 4;
  int l15  = lane & 15;
  int tm = blockIdx.x, tn = blockIdx.y;
  int row0 = tm * 32 + l15, row1 = row0 + 16;
  int col0 = tn * 32 + l15, col1 = col0 + 16;
  int Mc = M - 1, Nc = Npad - 1;
  int r0 = row0 > Mc ? Mc : row0;
  int r1 = row1 > Mc ? Mc : row1;
  int c0 = col0 > Nc ? Nc : col0;
  int c1 = col1 > Nc ? Nc : col1;
  int sub = half * 8;
  const _Float16* xp0 = X + (size_t)r0 * K + sub;
  const _Float16* xp1 = X + (size_t)r1 * K + sub;
  const _Float16* wp0 = W + (size_t)c0 * K + sub;
  const _Float16* wp1 = W + (size_t)c1 * K + sub;
  v8f acc00 = {}, acc01 = {}, acc10 = {}, acc11 = {};
  for (int kb = 0; kb < K; kb += 32) {
    v16h a0 = ldfrag(xp0 + kb);
    v16h a1 = ldfrag(xp1 + kb);
    v16h b0 = ldfrag(wp0 + kb);
    v16h b1 = ldfrag(wp1 + kb);
    __builtin_prefetch(xp0 + kb + 64, 0, 3);  // global_prefetch_b8 on streaming A
    __builtin_prefetch(xp1 + kb + 64, 0, 3);
    acc00 = __builtin_amdgcn_wmma_f32_16x16x32_f16(false, a0, false, b0, (short)0, acc00, false, false);
    acc01 = __builtin_amdgcn_wmma_f32_16x16x32_f16(false, a0, false, b1, (short)0, acc01, false, false);
    acc10 = __builtin_amdgcn_wmma_f32_16x16x32_f16(false, a1, false, b0, (short)0, acc10, false, false);
    acc11 = __builtin_amdgcn_wmma_f32_16x16x32_f16(false, a1, false, b1, (short)0, acc11, false, false);
  }
  // D layout: lane -> col = tile_n*16 + (l&15); rows = tile_m*16 + (l>>4)*8 + r
  float bb0 = (float)Bv[c0];
  float bb1 = (float)Bv[c1];
  int mb = tm * 32 + half * 8;
#pragma unroll
  for (int r = 0; r < 8; r++) {
    float v00 = acc00[r] + bb0, v01 = acc01[r] + bb1;
    float v10 = acc10[r] + bb0, v11 = acc11[r] + bb1;
    if (relu) {
      v00 = fmaxf(v00, 0.f); v01 = fmaxf(v01, 0.f);
      v10 = fmaxf(v10, 0.f); v11 = fmaxf(v11, 0.f);
    }
    int m0 = mb + r, m1 = mb + 16 + r;
    if (Yh) {
      if (m0 < M && col0 < N) Yh[(size_t)m0 * ldy + col0] = (_Float16)v00;
      if (m0 < M && col1 < N) Yh[(size_t)m0 * ldy + col1] = (_Float16)v01;
      if (m1 < M && col0 < N) Yh[(size_t)m1 * ldy + col0] = (_Float16)v10;
      if (m1 < M && col1 < N) Yh[(size_t)m1 * ldy + col1] = (_Float16)v11;
    } else {
      if (m0 < M && col0 < N) Yf[(size_t)m0 * ldy + col0] = v00;
      if (m0 < M && col1 < N) Yf[(size_t)m0 * ldy + col1] = v01;
      if (m1 < M && col0 < N) Yf[(size_t)m1 * ldy + col0] = v10;
      if (m1 < M && col1 < N) Yf[(size_t)m1 * ldy + col1] = v11;
    }
  }
}

// ---------------- max-pool over k rows per group, write at channel offset ----------------
__global__ void maxpool_k(const _Float16* __restrict__ Yin, int G, int k, int C,
                          _Float16* __restrict__ out, int ldo) {
  int i = blockIdx.x * blockDim.x + threadIdx.x;
  if (i >= G * C) return;
  int g = i / C, c = i % C;
  const _Float16* p = Yin + ((size_t)g * k) * C + c;
  float m = -1e30f;
  for (int s = 0; s < k; s++) m = fmaxf(m, (float)p[(size_t)s * C]);
  out[(size_t)g * ldo + c] = (_Float16)m;
}

// =======================================================================
extern "C" void kernel_launch(void* const* d_in, const int* in_sizes, int n_in,
                              void* d_out, int out_size, void* d_ws, size_t ws_size,
                              hipStream_t stream) {
  (void)in_sizes; (void)n_in; (void)out_size; (void)ws_size;
  const int B = 8, N1 = 8192;

  // ---- workspace carve (deterministic per call) ----
  size_t off = 0;
  auto carve = [&](size_t bytes) -> void* {
    void* p = (char*)d_ws + off;
    off = (off + bytes + 255) & ~(size_t)255;
    return p;
  };
  float*    xyzT  = (float*)carve((size_t)B * N1 * 3 * 4);
  float*    dist  = (float*)carve((size_t)B * N1 * 4);
  int*      fidx1 = (int*)carve((size_t)B * 512 * 4);
  float*    nxyz1 = (float*)carve((size_t)B * 512 * 3 * 4);
  int*      fidx2 = (int*)carve((size_t)B * 128 * 4);
  float*    nxyz2 = (float*)carve((size_t)B * 128 * 3 * 4);
  int*      gidx  = (int*)carve((size_t)B * 512 * 128 * 4);
  _Float16* l1pts = (_Float16*)carve((size_t)B * 512 * 320 * 2);
  _Float16* l2pts = (_Float16*)carve((size_t)B * 128 * 640 * 2);
  _Float16* sa3o  = (_Float16*)carve((size_t)B * 128 * 1024 * 2);
  _Float16* feat  = (_Float16*)carve((size_t)16 * 1024 * 2);
  _Float16* fc1o  = (_Float16*)carve((size_t)16 * 512 * 2);
  _Float16* fc2o  = (_Float16*)carve((size_t)16 * 256 * 2);
  _Float16* bufA  = (_Float16*)carve((size_t)524288 * 96 * 2);
  _Float16* bufB  = (_Float16*)carve((size_t)524288 * 128 * 2);

  // ---- fold all params (dict order: sa1[3][3], sa2[3][3], sa3[3], fc1, fc2, fc3) ----
  int cur = 1;
  struct Layer { const _Float16* w; const _Float16* b; int Ci, Kpad, Co, Npad; };
  auto fold = [&](int Ci, int Co, bool bn) -> Layer {
    const float* w  = (const float*)d_in[cur++];
    const float* bb = (const float*)d_in[cur++];
    const float* g = nullptr; const float* be = nullptr;
    if (bn) { g = (const float*)d_in[cur++]; be = (const float*)d_in[cur++]; }
    int Kpad = (Ci + 31) / 32 * 32;
    int Npad = (Co + 15) / 16 * 16;
    _Float16* w16 = (_Float16*)carve((size_t)Npad * Kpad * 2);
    _Float16* b16 = (_Float16*)carve((size_t)Npad * 2);
    int total = Npad * Kpad;
    fold_bn_k<<<(total + 255) / 256, 256, 0, stream>>>(w, bb, g, be, Co, Ci, Kpad, Npad, w16, b16);
    return Layer{w16, b16, Ci, Kpad, Co, Npad};
  };
  const int sa1w[3][3] = {{32, 32, 64}, {64, 64, 128}, {64, 96, 128}};
  const int sa2w[3][3] = {{64, 64, 128}, {128, 128, 256}, {128, 128, 256}};
  const int sa3w[3] = {256, 512, 1024};
  Layer L1[3][3], L2[3][3], L3[3];
  for (int br = 0; br < 3; br++) { int ci = 3;   for (int l = 0; l < 3; l++) { L1[br][l] = fold(ci, sa1w[br][l], true); ci = sa1w[br][l]; } }
  for (int br = 0; br < 3; br++) { int ci = 323; for (int l = 0; l < 3; l++) { L2[br][l] = fold(ci, sa2w[br][l], true); ci = sa2w[br][l]; } }
  { int ci = 643; for (int l = 0; l < 3; l++) { L3[l] = fold(ci, sa3w[l], true); ci = sa3w[l]; } }
  Layer Lf1 = fold(1024, 512, true);
  Layer Lf2 = fold(512, 256, true);
  Layer Lf3 = fold(256, 10, false);

  auto gemm = [&](const _Float16* X, const Layer& L, int M,
                  _Float16* Yh, float* Yf, int ldy, bool relu) {
    dim3 grid((M + 31) / 32, (L.Npad + 31) / 32);
    gemm_bias_act_k<<<grid, 32, 0, stream>>>(X, L.w, L.b, Yh, Yf,
                                             M, L.Kpad, L.Co, L.Npad, ldy, relu ? 1 : 0);
  };

  auto run_branch = [&](const float* lvl_xyz, const float* nxyz, const _Float16* ftr, int Cf,
                        int N, int np, float radius, int k, Layer* Ls,
                        _Float16* outPts, int outCld, int choff) {
    int G = B * np;
    query_ball_k<<<G, 32, 0, stream>>>(lvl_xyz, nxyz, N, np, radius * radius, k, gidx);
    int rows = G * k;
    gather_k<<<(rows + 255) / 256, 256, 0, stream>>>(lvl_xyz, nxyz, gidx, ftr, Cf, N, np, k,
                                                     Ls[0].Kpad, rows, bufA);
    _Float16* pa = bufA; _Float16* pb = bufB;
    for (int l = 0; l < 3; l++) {
      gemm(pa, Ls[l], rows, pb, nullptr, Ls[l].Co, true);
      _Float16* t = pa; pa = pb; pb = t;
    }
    int C = Ls[2].Co;
    maxpool_k<<<(G * C + 255) / 256, 256, 0, stream>>>(pa, G, k, C, outPts + choff, outCld);
  };

  // ---- pipeline ----
  const float* xyz_in = (const float*)d_in[0];
  transpose_xyz_k<<<(B * N1 * 3 + 255) / 256, 256, 0, stream>>>(xyz_in, xyzT, B, N1);

  fps_k<<<B, THREADS, 0, stream>>>(xyzT, N1, 512, fidx1, nxyz1, dist);
  run_branch(xyzT, nxyz1, nullptr, 0, N1, 512, 0.1f, 16,  L1[0], l1pts, 320, 0);
  run_branch(xyzT, nxyz1, nullptr, 0, N1, 512, 0.2f, 32,  L1[1], l1pts, 320, 64);
  run_branch(xyzT, nxyz1, nullptr, 0, N1, 512, 0.4f, 128, L1[2], l1pts, 320, 192);

  fps_k<<<B, THREADS, 0, stream>>>(nxyz1, 512, 128, fidx2, nxyz2, dist);
  run_branch(nxyz1, nxyz2, l1pts, 320, 512, 128, 0.2f, 32,  L2[0], l2pts, 640, 0);
  run_branch(nxyz1, nxyz2, l1pts, 320, 512, 128, 0.4f, 64,  L2[1], l2pts, 640, 128);
  run_branch(nxyz1, nxyz2, l1pts, 320, 512, 128, 0.8f, 128, L2[2], l2pts, 640, 384);

  // SA3: 1024 rows of 643 -> 256 -> 512 -> 1024, max over 128 points per batch
  int rows3 = B * 128;
  build_sa3_k<<<(rows3 + 255) / 256, 256, 0, stream>>>(nxyz2, l2pts, 640, L3[0].Kpad, rows3, bufA);
  gemm(bufA, L3[0], rows3, bufB, nullptr, 256, true);
  gemm(bufB, L3[1], rows3, bufA, nullptr, 512, true);
  gemm(bufA, L3[2], rows3, sa3o, nullptr, 1024, true);
  maxpool_k<<<(B * 1024 + 255) / 256, 256, 0, stream>>>(sa3o, B, 128, 1024, feat, 1024);

  // FC head
  gemm(feat, Lf1, B, fc1o, nullptr, 512, true);
  gemm(fc1o, Lf2, B, fc2o, nullptr, 256, true);
  gemm(fc2o, Lf3, B, nullptr, (float*)d_out, 10, false);
}
